// Resnet34ExtendTorch_420906795303
// MI455X (gfx1250) — matrix operand order repaired and checked by reading, not compiled, not run on hardware
//
#include <hip/hip_runtime.h>
#include <cstdint>
#include <cstddef>

typedef _Float16 h16;
typedef __attribute__((ext_vector_type(16))) _Float16 v16h;
typedef __attribute__((ext_vector_type(8)))  _Float16 v8h;
typedef __attribute__((ext_vector_type(8)))  float    v8f;

union U16h { v16h v; v8h h[2]; };

#define EPSBN 1e-5f

// ---------------------------------------------------------------------------
// Fold BN into conv weights: OIHW fp32 -> [O][K] f16 with K = (kh*KW+kw)*Ci+ci
// bias[o] = bn.b - bn.m * scale   (scale = bn.g * rsqrt(bn.v + eps))
// ---------------------------------------------------------------------------
__global__ __launch_bounds__(256) void fold_kernel(
    const float* __restrict__ w,
    const float* __restrict__ g, const float* __restrict__ bb,
    const float* __restrict__ bm, const float* __restrict__ bv,
    h16* __restrict__ wt, float* __restrict__ bias, int Co, int Ci, int KH)
{
  int K = KH * KH * Ci;
  long long tot = (long long)Co * K;
  long long idx = (long long)blockIdx.x * 256 + threadIdx.x;
  if (idx >= tot) return;
  int o = (int)(idx / K);
  int k = (int)(idx % K);
  int tap = k / Ci, ci = k - tap * Ci;
  int kh = tap / KH, kw = tap - kh * KH;
  float scale = g[o] * rsqrtf(bv[o] + EPSBN);
  wt[idx] = (h16)(w[(((long long)o * Ci + ci) * KH + kh) * KH + kw] * scale);
  if (k == 0) bias[o] = bb[o] - bm[o] * scale;
}

// ---------------------------------------------------------------------------
// Stem: conv 1->3 k3 p1 s1 + conv-bias + BN + ReLU.  X NCHW (C=1) fp32 ->
// NHWC f16 (8,512,512,3)
// ---------------------------------------------------------------------------
__global__ __launch_bounds__(256) void stem_kernel(
    const float* __restrict__ X, const float* __restrict__ w,
    const float* __restrict__ cb,
    const float* __restrict__ g, const float* __restrict__ b,
    const float* __restrict__ m, const float* __restrict__ v,
    h16* __restrict__ out)
{
  int idx = blockIdx.x * 256 + threadIdx.x;         // over 8*512*512 pixels
  int ww = idx & 511, hh = (idx >> 9) & 511, n = idx >> 18;
  float ac[3] = {0.f, 0.f, 0.f};
  const float* Xp = X + (size_t)n * 512 * 512;
  for (int kh = 0; kh < 3; kh++) {
    int ih = hh - 1 + kh; if ((unsigned)ih >= 512u) continue;
    for (int kw = 0; kw < 3; kw++) {
      int iw = ww - 1 + kw; if ((unsigned)iw >= 512u) continue;
      float xv = Xp[ih * 512 + iw];
      ac[0] += xv * w[0 * 9 + kh * 3 + kw];
      ac[1] += xv * w[1 * 9 + kh * 3 + kw];
      ac[2] += xv * w[2 * 9 + kh * 3 + kw];
    }
  }
  size_t ob = (size_t)idx * 3;
  for (int c = 0; c < 3; c++) {
    float sc = g[c] * rsqrtf(v[c] + EPSBN);
    float sh = b[c] - m[c] * sc;
    float val = (ac[c] + cb[c]) * sc + sh;
    out[ob + c] = (h16)fmaxf(val, 0.f);
  }
}

// ---------------------------------------------------------------------------
// conv1: 3->64 k7 s2 p3 + folded BN + ReLU. NHWC f16 in (8,512,512,3) ->
// (8,256,256,64). Weights [64][147] staged in LDS.
// ---------------------------------------------------------------------------
__global__ __launch_bounds__(256) void conv7_kernel(
    const h16* __restrict__ in, const h16* __restrict__ wt,
    const float* __restrict__ bias, h16* __restrict__ out)
{
  __shared__ h16 Wsh[64 * 147];
  for (int i = threadIdx.x; i < 64 * 147; i += 256) Wsh[i] = wt[i];
  __syncthreads();
  size_t idx = (size_t)blockIdx.x * 256 + threadIdx.x;   // 8*256*256*64
  int c = (int)(idx & 63); int pix = (int)(idx >> 6);
  int ww = pix & 255, hh = (pix >> 8) & 255, n = pix >> 16;
  float s = bias[c];
  const h16* wp = Wsh + c * 147;
  for (int kh = 0; kh < 7; kh++) {
    int ih = hh * 2 - 3 + kh; if ((unsigned)ih >= 512u) continue;
    for (int kw = 0; kw < 7; kw++) {
      int iw = ww * 2 - 3 + kw; if ((unsigned)iw >= 512u) continue;
      const h16* ip = in + ((size_t)(n * 512 + ih) * 512 + iw) * 3;
      int t = (kh * 7 + kw) * 3;
      s += (float)ip[0] * (float)wp[t]
         + (float)ip[1] * (float)wp[t + 1]
         + (float)ip[2] * (float)wp[t + 2];
    }
  }
  out[idx] = (h16)fmaxf(s, 0.f);
}

// ---------------------------------------------------------------------------
// MaxPool 3x3 s2 p1: (8,256,256,64) -> (8,128,128,64) NHWC f16
// ---------------------------------------------------------------------------
__global__ __launch_bounds__(256) void maxpool_kernel(
    const h16* __restrict__ in, h16* __restrict__ out)
{
  size_t idx = (size_t)blockIdx.x * 256 + threadIdx.x;   // 8*128*128*64
  int c = (int)(idx & 63); int pix = (int)(idx >> 6);
  int ww = pix & 127, hh = (pix >> 7) & 127, n = pix >> 14;
  float mx = -3.4e38f;
  for (int kh = 0; kh < 3; kh++) {
    int ih = hh * 2 - 1 + kh; if ((unsigned)ih >= 256u) continue;
    for (int kw = 0; kw < 3; kw++) {
      int iw = ww * 2 - 1 + kw; if ((unsigned)iw >= 256u) continue;
      mx = fmaxf(mx, (float)in[((size_t)(n * 256 + ih) * 256 + iw) * 64 + c]);
    }
  }
  out[idx] = (h16)mx;
}

// ---------------------------------------------------------------------------
// Implicit-GEMM conv via WMMA (f16 in, f32 acc), 2x2 fragment blocking,
// LDS double-buffered software pipeline.
// Block: 256 thr = 8 waves (4 M-waves x 2 N-waves), each wave computes a
// 32x32 output tile (4 accumulators) -> block tile 128(M) x 64(N).
// K staged 64/iter; stage s+1's global loads are issued before computing
// stage s (one barrier per stage, latency hidden behind 8 WMMAs/wave).
// NHWC f16 act, [Co][K] f16 weights (BN-folded).
// Epilogue: +bias, +residual (optional), ReLU (optional), f16 NHWC store,
// optional fp32 NCHW store (final feature output).
// Requires: Ci % 64 == 0, Co % 64 == 0, (8*Hout*Wout) % 128 == 0.
// ---------------------------------------------------------------------------
__global__ __launch_bounds__(256) void conv_wmma_kernel(
    const h16* __restrict__ in, const h16* __restrict__ wt,
    const float* __restrict__ bias, const h16* res,
    h16* out, float* out_f32,
    int Hin, int Win, int Ci, int Hout, int Wout, int Co,
    int KH, int pad, int stride, int do_relu)
{
  __shared__ h16 Ash[2][128][72];
  __shared__ h16 Bsh[2][64][72];
  const int K = KH * KH * Ci;
  const int KSTAGES = K >> 6;
  const int tid = threadIdx.x;
  const int blockM = blockIdx.x * 128;
  const int blockN = blockIdx.y * 64;
  const int wid = tid >> 5, lane = tid & 31;
  const int wm = wid & 3, wn = wid >> 2;   // wave tile: rows wm*32.., cols wn*32..
  // A-tile loader: 128 rows x 64 k, 32 halves (4x b128) per thread
  const int ar = tid >> 1;
  const int ak = (tid & 1) * 32;
  // B-tile loader: 64 cols x 64 k, 16 halves (2x b128) per thread
  const int bc = tid >> 2;
  const int bk = (tid & 3) * 16;
  const int HW = Hout * Wout;

  int m = blockM + ar;
  int n_img = m / HW; int rm = m - n_img * HW;
  int oh = rm / Wout, ow = rm - (rm / Wout) * Wout;

  v8f acc00 = {}, acc01 = {}, acc10 = {}, acc11 = {};
  const int arow0 = wm * 32 + (lane & 15);
  const int arow1 = arow0 + 16;
  const int kbA = (lane >> 4) * 16;       // A: lanes>=16 hold K=16..31
  const int bcol0 = wn * 32 + (lane & 15);
  const int bcol1 = bcol0 + 16;
  const int kbB = (lane >> 4) * 16;       // B: lanes>=16 hold K=16..31

  const h16* wrow = wt + (size_t)(blockN + bc) * K;

  uint4 areg[4], breg[2];

  // global fetch of one 64-wide K stage into registers
  auto load_stage = [&](int K0) {
    int k = K0 + ak;
    int tap = k / Ci, c0 = k - tap * Ci;
    int kh = tap / KH, kw = tap - kh * KH;
    int ih = oh * stride - pad + kh, iw = ow * stride - pad + kw;
    areg[0] = make_uint4(0u, 0u, 0u, 0u);
    areg[1] = areg[0]; areg[2] = areg[0]; areg[3] = areg[0];
    if ((unsigned)ih < (unsigned)Hin && (unsigned)iw < (unsigned)Win) {
      const uint4* p = (const uint4*)(in + ((size_t)(n_img * Hin + ih) * Win + iw) * Ci + c0);
      areg[0] = p[0]; areg[1] = p[1]; areg[2] = p[2]; areg[3] = p[3];
    }
    const uint4* bp = (const uint4*)(wrow + K0 + bk);
    breg[0] = bp[0]; breg[1] = bp[1];
    if (K0 + 128 < K) __builtin_prefetch((const void*)(bp + 16), 0, 1);
  };
  // commit staged registers into LDS buffer `buf`
  auto store_stage = [&](int buf) {
    *(uint4*)&Ash[buf][ar][ak]      = areg[0];
    *(uint4*)&Ash[buf][ar][ak + 8]  = areg[1];
    *(uint4*)&Ash[buf][ar][ak + 16] = areg[2];
    *(uint4*)&Ash[buf][ar][ak + 24] = areg[3];
    *(uint4*)&Bsh[buf][bc][bk]      = breg[0];
    *(uint4*)&Bsh[buf][bc][bk + 8]  = breg[1];
  };

  load_stage(0);
  store_stage(0);

  for (int s = 0; s < KSTAGES; s++) {
    const int buf = s & 1;
    const bool has_next = (s + 1 < KSTAGES);
    if (has_next) load_stage((s + 1) << 6);   // overlap with this stage's math
    __syncthreads();                          // LDS[buf] visible to all waves
    // ---- 8 WMMAs per 64-wide K stage with 2x2 fragment reuse
#pragma unroll
    for (int ks = 0; ks < 64; ks += 32) {
      U16h af0, af1, bf0, bf1;
      af0.h[0] = *(const v8h*)&Ash[buf][arow0][ks + kbA];
      af0.h[1] = *(const v8h*)&Ash[buf][arow0][ks + kbA + 8];
      af1.h[0] = *(const v8h*)&Ash[buf][arow1][ks + kbA];
      af1.h[1] = *(const v8h*)&Ash[buf][arow1][ks + kbA + 8];
      bf0.h[0] = *(const v8h*)&Bsh[buf][bcol0][ks + kbB];
      bf0.h[1] = *(const v8h*)&Bsh[buf][bcol0][ks + kbB + 8];
      bf1.h[0] = *(const v8h*)&Bsh[buf][bcol1][ks + kbB];
      bf1.h[1] = *(const v8h*)&Bsh[buf][bcol1][ks + kbB + 8];
      acc00 = __builtin_amdgcn_wmma_f32_16x16x32_f16(
          false, af0.v, false, bf0.v, (short)0, acc00, false, false);
      acc01 = __builtin_amdgcn_wmma_f32_16x16x32_f16(
          false, af0.v, false, bf1.v, (short)0, acc01, false, false);
      acc10 = __builtin_amdgcn_wmma_f32_16x16x32_f16(
          false, af1.v, false, bf0.v, (short)0, acc10, false, false);
      acc11 = __builtin_amdgcn_wmma_f32_16x16x32_f16(
          false, af1.v, false, bf1.v, (short)0, acc11, false, false);
    }
    // write next stage into the other buffer; made visible by next barrier
    if (has_next) store_stage(buf ^ 1);
  }

  // ---- epilogue: C/D layout: lane<16 -> M=j, lane>=16 -> M=j+8; N=lane&15
  const int nloc = lane & 15;
  const int mb = blockM + wm * 32 + ((lane >> 4) * 8);
  const int co0 = blockN + wn * 32 + nloc;
  const int co1 = co0 + 16;
  const float bv0 = bias[co0];
  const float bv1 = bias[co1];

  auto store_tile = [&](const v8f& a, int mi, int co, float bv) {
#pragma unroll
    for (int j = 0; j < 8; j++) {
      int mm = mb + mi * 16 + j;
      float v = a[j] + bv;
      if (res) v += (float)res[(size_t)mm * Co + co];
      if (do_relu) v = fmaxf(v, 0.f);
      out[(size_t)mm * Co + co] = (h16)v;
      if (out_f32) {
        int ni = mm / HW; int r2 = mm - ni * HW;
        int hh = r2 / Wout, ww = r2 - hh * Wout;
        out_f32[(((size_t)ni * Co + co) * Hout + hh) * Wout + ww] = v;
      }
    }
  };
  store_tile(acc00, 0, co0, bv0);
  store_tile(acc01, 0, co1, bv1);
  store_tile(acc10, 1, co0, bv0);
  store_tile(acc11, 1, co1, bv1);
}

// ---------------------------------------------------------------------------
// CAM: per (b, class) block; 256 threads == 256 pixels (16x16). Dot over 512
// channels, ReLU, block max-reduce, normalize, write fp32 cam.
// ---------------------------------------------------------------------------
__global__ __launch_bounds__(256) void cam_kernel(
    const h16* __restrict__ feat, const float* __restrict__ clsw,
    float* __restrict__ cam_out, int HW, int C, int NC)
{
  int b = blockIdx.x / NC, n = blockIdx.x % NC;
  int p = threadIdx.x;
  const h16* f = feat + ((size_t)b * HW + p) * C;
  const float* wv = clsw + (size_t)n * C;
  float s = 0.f;
  for (int d = 0; d < C; d += 8) {
    v8h hv = *(const v8h*)(f + d);
#pragma unroll
    for (int q = 0; q < 8; q++) s += (float)hv[q] * wv[d + q];
  }
  s = fmaxf(s, 0.f);
  __shared__ float red[256];
  red[p] = s; __syncthreads();
  for (int stn = 128; stn > 0; stn >>= 1) {
    if (p < stn) red[p] = fmaxf(red[p], red[p + stn]);
    __syncthreads();
  }
  cam_out[((size_t)b * NC + n) * HW + p] = s / (red[0] + 1e-5f);
}

// ---------------------------------------------------------------------------
// cam_feature[b,n,d] = sum_hw cam[b,n,hw] * feat[b,hw,d] / HW
// ---------------------------------------------------------------------------
__global__ __launch_bounds__(256) void camfeat_kernel(
    const h16* __restrict__ feat, const float* __restrict__ cam,
    float* __restrict__ outp, int HW, int C, int NC)
{
  int b = blockIdx.x / NC, n = blockIdx.x % NC;
  __shared__ float cs[256];
  cs[threadIdx.x] = cam[((size_t)b * NC + n) * HW + threadIdx.x];
  __syncthreads();
  for (int d = threadIdx.x; d < C; d += 256) {
    float s = 0.f;
    for (int p = 0; p < HW; p++)
      s += cs[p] * (float)feat[((size_t)b * HW + p) * C + d];
    outp[((size_t)b * NC + n) * C + d] = s / (float)HW;
  }
}

// ---------------------------------------------------------------------------
// Host orchestration
// ---------------------------------------------------------------------------
struct BNp { const float *b, *g, *m, *v; };

extern "C" void kernel_launch(void* const* d_in, const int* in_sizes, int n_in,
                              void* d_out, int out_size, void* d_ws, size_t ws_size,
                              hipStream_t stream) {
  (void)in_sizes; (void)n_in; (void)out_size; (void)ws_size;
  auto F = [&](int i) -> const float* { return (const float*)d_in[i]; };

  // --- bind inputs in JAX pytree flat order (dict keys sorted) ---
  int ip = 0;
  const float* X = F(ip++);                                    // 'X'
  BNp bn1 = { F(ip), F(ip + 1), F(ip + 2), F(ip + 3) }; ip += 4; // params.bn1 {b,g,m,v}
  const float* cls_w   = F(ip++);                              // params.cls_w
  const float* conv1_w = F(ip++);                              // params.conv1_w

  struct Blk { BNp bn1, bn2, dbn; const float *dw, *w1, *w2; int hasd, ci, co, stride; };
  Blk blks[16]; int nb = 0;
  const int LAY[4][3] = { {64, 3, 1}, {128, 4, 2}, {256, 6, 2}, {512, 3, 2} };
  int in_c = 64;
  for (int l = 0; l < 4; l++) {
    int c = LAY[l][0], nblk = LAY[l][1], s = LAY[l][2];
    for (int bi = 0; bi < nblk; bi++) {
      int st = (bi == 0) ? s : 1;
      Blk bk{};
      bk.ci = in_c; bk.co = c; bk.stride = st;
      bk.hasd = (st != 1 || in_c != c) ? 1 : 0;
      bk.bn1 = { F(ip), F(ip + 1), F(ip + 2), F(ip + 3) }; ip += 4;
      bk.bn2 = { F(ip), F(ip + 1), F(ip + 2), F(ip + 3) }; ip += 4;
      if (bk.hasd) {
        bk.dbn = { F(ip), F(ip + 1), F(ip + 2), F(ip + 3) }; ip += 4;
        bk.dw = F(ip++);
      } else bk.dw = nullptr;
      bk.w1 = F(ip++);
      bk.w2 = F(ip++);
      blks[nb++] = bk;
      in_c = c;
    }
  }
  const float* stem_b = F(ip++);                               // params.stem_b
  BNp sbn = { F(ip), F(ip + 1), F(ip + 2), F(ip + 3) }; ip += 4; // params.stem_bn
  const float* stem_w = F(ip++);                               // params.stem_w

  // --- workspace bump allocator ---
  char* wsb = (char*)d_ws; size_t off = 0;
  auto alloc = [&](size_t bytes) -> void* {
    void* p = wsb + off; off += (bytes + 255) & ~(size_t)255; return p;
  };

  h16* actStem = (h16*)alloc((size_t)8 * 512 * 512 * 3 * sizeof(h16));  // 12.6 MB
  h16* bufY    = (h16*)alloc((size_t)8 * 256 * 256 * 64 * sizeof(h16)); // 67 MB (conv1 out / y1)
  h16* cur     = (h16*)alloc((size_t)8 * 128 * 128 * 64 * sizeof(h16)); // 16.8 MB (x / out)
  h16* bufI    = (h16*)alloc((size_t)8 * 64 * 64 * 128 * sizeof(h16));  // 8.4 MB (identity)

  auto fold = [&](const float* w, BNp bn, int Co, int Ci, int KH, h16*& wtp, float*& bp) {
    wtp = (h16*)alloc((size_t)Co * KH * KH * Ci * sizeof(h16));
    bp  = (float*)alloc((size_t)Co * sizeof(float));
    long long tot = (long long)Co * KH * KH * Ci;
    int blocks = (int)((tot + 255) / 256);
    fold_kernel<<<blocks, 256, 0, stream>>>(w, bn.g, bn.b, bn.m, bn.v, wtp, bp, Co, Ci, KH);
  };

  h16* wt_c1; float* b_c1;
  fold(conv1_w, bn1, 64, 3, 7, wt_c1, b_c1);

  struct BW { h16 *w1, *w2, *dw; float *b1, *b2, *db; } bw[16];
  for (int i = 0; i < nb; i++) {
    fold(blks[i].w1, blks[i].bn1, blks[i].co, blks[i].ci, 3, bw[i].w1, bw[i].b1);
    fold(blks[i].w2, blks[i].bn2, blks[i].co, blks[i].co, 3, bw[i].w2, bw[i].b2);
    if (blks[i].hasd) fold(blks[i].dw, blks[i].dbn, blks[i].co, blks[i].ci, 1, bw[i].dw, bw[i].db);
    else { bw[i].dw = nullptr; bw[i].db = nullptr; }
  }

  // --- forward pass ---
  stem_kernel<<<(8 * 512 * 512) / 256, 256, 0, stream>>>(
      X, stem_w, stem_b, sbn.g, sbn.b, sbn.m, sbn.v, actStem);
  conv7_kernel<<<(8 * 256 * 256 * 64) / 256, 256, 0, stream>>>(actStem, wt_c1, b_c1, bufY);
  maxpool_kernel<<<(8 * 128 * 128 * 64) / 256, 256, 0, stream>>>(bufY, cur);

  auto launch_conv = [&](const h16* in, const h16* wtp, const float* bp, const h16* resp,
                         h16* outp, float* of32, int Hin, int Win, int Ci,
                         int Ho, int Wo, int Co, int KH, int pad, int st, int relu) {
    int M = 8 * Ho * Wo;
    dim3 g(M / 128, Co / 64);
    conv_wmma_kernel<<<g, 256, 0, stream>>>(in, wtp, bp, resp, outp, of32,
                                            Hin, Win, Ci, Ho, Wo, Co, KH, pad, st, relu);
  };

  float* outF = (float*)d_out;
  int H = 128, W = 128;
  for (int i = 0; i < nb; i++) {
    int st = blks[i].stride, Ci = blks[i].ci, Co = blks[i].co;
    int Ho = H / st, Wo = W / st;
    // y1 = relu(bn1(conv3x3(x)))
    launch_conv(cur, bw[i].w1, bw[i].b1, nullptr, bufY, nullptr,
                H, W, Ci, Ho, Wo, Co, 3, 1, st, 1);
    const h16* resp = cur;
    if (blks[i].hasd) {
      // idn = dbn(conv1x1(x))
      launch_conv(cur, bw[i].dw, bw[i].db, nullptr, bufI, nullptr,
                  H, W, Ci, Ho, Wo, Co, 1, 0, st, 0);
      resp = bufI;
    }
    // out = relu(bn2(conv3x3(y1)) + idn); final block also emits fp32 feature
    float* f32p = (i == nb - 1) ? outF : nullptr;
    launch_conv(bufY, bw[i].w2, bw[i].b2, resp, cur, f32p,
                Ho, Wo, Co, Ho, Wo, Co, 3, 1, 1, 1);
    H = Ho; W = Wo;
  }

  // --- CAM head ---
  float* cf_out  = outF + (size_t)8 * 512 * 16 * 16;   // cam_feature (8,20,512)
  float* cam_out = cf_out + (size_t)8 * 20 * 512;      // cam (8,20,16,16)
  cam_kernel<<<8 * 20, 256, 0, stream>>>(cur, cls_w, cam_out, 256, 512, 20);
  camfeat_kernel<<<8 * 20, 256, 0, stream>>>(cur, cam_out, cf_out, 256, 512, 20);
}
